// SelfDistil_W2V2BASE_AASISTL_6674379178618
// MI455X (gfx1250) — matrix operand order repaired and checked by reading, not compiled, not run on hardware
//
#include <hip/hip_runtime.h>
#include <math.h>

// ---------------------------------------------------------------------------
// AASIST-style head, MI455X (gfx1250, wave32).
// Heavy GEMM/conv work on v_wmma_f32_16x16x32_f16 (f32 accumulate).
// Round 2: branch-free fragment gathers (clamped unconditional loads),
// f16 zero-padded weight staging (aligned v8h B-fragment loads), LDS-resident
// im2col decode table.
// Input pointer order assumption: depth-first INSERTION-ORDER flattening of
// the setup_inputs() pytree.
// ---------------------------------------------------------------------------

typedef __attribute__((ext_vector_type(16))) _Float16 v16h;
typedef __attribute__((ext_vector_type(8)))  _Float16 v8h;
typedef __attribute__((ext_vector_type(8)))  float    v8f;

#define RSQ 0.9999950000374996f  // 1/sqrt(1+1e-5)  (BN running var = 1)

__device__ __forceinline__ float selu_f(float x) {
  const float a = 1.6732632423543772f, s = 1.0507009873554805f;
  return x > 0.f ? s * x : s * a * (expf(x) - 1.f);
}

// ---------------- weight staging ----------------
// LL weight (K,N) f32 -> transposed f16 (N,K)
__global__ void k_prep_wt(const float* __restrict__ W, _Float16* __restrict__ WT,
                          int K, int N) {
  int idx = blockIdx.x * blockDim.x + threadIdx.x;
  if (idx >= K * N) return;
  int n = idx / K, k = idx - n * K;
  WT[idx] = (_Float16)W[(size_t)k * N + n];
}

// conv weight (Co,K) f32 (OIHW is row-major over K=Ci*kh*kw) -> zero-padded f16 (Co16,K32)
__global__ void k_prep_w(const float* __restrict__ W, _Float16* __restrict__ WP,
                         int Co, int K, int Co16, int K32) {
  int idx = blockIdx.x * blockDim.x + threadIdx.x;
  if (idx >= Co16 * K32) return;
  int n = idx / K32, k = idx - n * K32;
  WP[idx] = (n < Co && k < K) ? (_Float16)W[(size_t)n * K + k] : (_Float16)0.f;
}

// im2col decode table: k -> {ci, dh(=khi-ph), dw(=kwi-pw), valid}
__global__ void k_prep_tab(int4* __restrict__ tab, int K, int K32,
                           int kh, int kw, int ph, int pw) {
  int k = blockIdx.x * blockDim.x + threadIdx.x;
  if (k >= K32) return;
  int khw = kh * kw;
  int kc = k < K ? k : 0;
  int ci = kc / khw; int r2 = kc - ci * khw; int khi = r2 / kw; int kwi = r2 - khi * kw;
  int4 t; t.x = ci; t.y = khi - ph; t.z = kwi - pw; t.w = (k < K) ? 1 : 0;
  tab[k] = t;
}

// ---------------- WMMA GEMM (M%16==0, N%16==0, K%32==0) ----------------
// Y[M,N] = A[M,K](f32) * WT[N,K](f16, pre-transposed) + bias
__global__ void k_gemm(const float* __restrict__ A, const _Float16* __restrict__ WT,
                       const float* __restrict__ bias, float* __restrict__ Y,
                       int M, int N, int K) {
  int gw   = (blockIdx.x * blockDim.x + threadIdx.x) >> 5;
  int lane = threadIdx.x & 31;
  int ntiles = N >> 4;
  int mt = gw / ntiles, nt = gw % ntiles;
  if (mt >= (M >> 4)) return;
  int half = lane >> 4, l = lane & 15;
  int mr = mt * 16 + l;
  int nc = nt * 16 + l;
  const float*    arow = A  + (size_t)mr * K;
  const _Float16* brow = WT + (size_t)nc * K;
  v8f acc = {};
  for (int k0 = 0; k0 < K; k0 += 32) {
    float4 a0 = *(const float4*)(arow + k0 + 8 * half);
    float4 a1 = *(const float4*)(arow + k0 + 8 * half + 4);
    float4 a2 = *(const float4*)(arow + k0 + 16 + 8 * half);
    float4 a3 = *(const float4*)(arow + k0 + 16 + 8 * half + 4);
    v8h b0 = *(const v8h*)(brow + k0 + 8 * half);
    v8h b1 = *(const v8h*)(brow + k0 + 16 + 8 * half);
    v16h a, b;
    a[0]  = (_Float16)a0.x; a[1]  = (_Float16)a0.y; a[2]  = (_Float16)a0.z; a[3]  = (_Float16)a0.w;
    a[4]  = (_Float16)a1.x; a[5]  = (_Float16)a1.y; a[6]  = (_Float16)a1.z; a[7]  = (_Float16)a1.w;
    a[8]  = (_Float16)a2.x; a[9]  = (_Float16)a2.y; a[10] = (_Float16)a2.z; a[11] = (_Float16)a2.w;
    a[12] = (_Float16)a3.x; a[13] = (_Float16)a3.y; a[14] = (_Float16)a3.z; a[15] = (_Float16)a3.w;
#pragma unroll
    for (int e = 0; e < 8; ++e) { b[e] = b0[e]; b[8 + e] = b1[e]; }
    acc = __builtin_amdgcn_wmma_f32_16x16x32_f16(false, a, false, b, (short)0,
                                                 acc, false, false);
  }
#pragma unroll
  for (int r = 0; r < 8; ++r) {
    int m = mt * 16 + r + 8 * half;
    int n = nt * 16 + l;
    Y[(size_t)m * N + n] = acc[r] + bias[n];
  }
}

// ------------- WMMA implicit-GEMM conv (NCHW), fused epilogues -------------
// M = B*Ho*Wo must be a multiple of 16 (true: B=128).
// pre  (per-Ci): x = selu(x*pres[ci]*RSQ + preb[ci])
// post (per-Co): y = selu(y)*posts[n]*RSQ + postb[n]
// addend: y += addend[same NCHW index]
__global__ void k_conv(const float* __restrict__ X, const _Float16* __restrict__ WP,
                       const int4* __restrict__ tab,
                       const float* __restrict__ bias, const float* __restrict__ addend,
                       const float* __restrict__ pres, const float* __restrict__ preb,
                       const float* __restrict__ posts, const float* __restrict__ postb,
                       float* __restrict__ Y,
                       int M, int Ci, int Hi, int Wi, int Co, int Ho, int Wo,
                       int K, int K32) {
  __shared__ int4 stab[192];
  for (int i = threadIdx.x; i < K32; i += blockDim.x) stab[i] = tab[i];
  __syncthreads();

  int gw   = (blockIdx.x * blockDim.x + threadIdx.x) >> 5;
  int lane = threadIdx.x & 31;
  int hw = Ho * Wo;
  int ntiles = (Co + 15) >> 4;
  int mt = gw / ntiles, nt = gw % ntiles;
  if (mt >= (M >> 4)) return;
  int half = lane >> 4, l = lane & 15;
  int row = mt * 16 + l;                      // always < M (M%16==0)
  int b_ = row / hw; int rr = row - b_ * hw; int oh = rr / Wo; int ow = rr - oh * Wo;
  int HiWi = Hi * Wi;
  const float* xb = X + (size_t)b_ * Ci * HiWi;
  int nc = nt * 16 + l;                       // < Co16; WP rows padded with zeros
  const _Float16* brow = WP + (size_t)nc * K32;
  v8f acc = {};
  for (int k0 = 0; k0 < K32; k0 += 32) {
    v16h a, b;
#pragma unroll
    for (int e = 0; e < 16; ++e) {
      int ka = k0 + (e & 7) + ((e >> 3) << 4) + 8 * half;
      int4 t = stab[ka];
      int ih = oh + t.y, iw = ow + t.z;
      bool inb = t.w && ((unsigned)ih < (unsigned)Hi) && ((unsigned)iw < (unsigned)Wi);
      int ihc = ih < 0 ? 0 : (ih >= Hi ? Hi - 1 : ih);
      int iwc = iw < 0 ? 0 : (iw >= Wi ? Wi - 1 : iw);
      float av = xb[(size_t)t.x * HiWi + ihc * Wi + iwc];   // unconditional, clamped
      if (pres) av = selu_f(av * pres[t.x] * RSQ + preb[t.x]);
      a[e] = inb ? (_Float16)av : (_Float16)0.f;
    }
    v8h b0 = *(const v8h*)(brow + k0 + 8 * half);
    v8h b1 = *(const v8h*)(brow + k0 + 16 + 8 * half);
#pragma unroll
    for (int e = 0; e < 8; ++e) { b[e] = b0[e]; b[8 + e] = b1[e]; }
    acc = __builtin_amdgcn_wmma_f32_16x16x32_f16(false, a, false, b, (short)0,
                                                 acc, false, false);
  }
#pragma unroll
  for (int r = 0; r < 8; ++r) {
    int m = mt * 16 + r + 8 * half;
    int n = nt * 16 + l;
    if (n < Co) {
      int b2 = m / hw; int rr2 = m - b2 * hw; int oh2 = rr2 / Wo; int ow2 = rr2 - oh2 * Wo;
      float v = acc[r] + bias[n];
      if (posts) v = selu_f(v) * posts[n] * RSQ + postb[n];
      size_t oidx = (((size_t)b2 * Co + n) * Ho + oh2) * Wo + ow2;
      if (addend) v += addend[oidx];
      Y[oidx] = v;
    }
  }
}

// -------- maxpool 3x3 stride 3 on xs viewed as (B,1,128,201) + BN(1) + SELU --
__global__ void k_pool_bnselu(const float* __restrict__ XS, const float* __restrict__ g,
                              const float* __restrict__ bb, float* __restrict__ Y,
                              int B, int H, int W, int T, int C) {
  int idx = blockIdx.x * blockDim.x + threadIdx.x;
  int n = B * H * W;
  if (idx >= n) return;
  int b = idx / (H * W); int rr = idx - b * H * W; int oh = rr / W; int ow = rr - oh * W;
  float m = -1e30f;
#pragma unroll
  for (int dh = 0; dh < 3; ++dh)
#pragma unroll
    for (int dw = 0; dw < 3; ++dw) {
      int t = 3 * ow + dw;   // width dim = time
      int c = 3 * oh + dh;   // height dim = feature
      m = fmaxf(m, XS[((size_t)b * T + t) * C + c]);
    }
  Y[idx] = selu_f(m * g[0] * RSQ + bb[0]);
}

// --------------------------- elementwise BN + SELU ---------------------------
__global__ void k_bnselu(float* __restrict__ X, const float* __restrict__ g,
                         const float* __restrict__ bb, long n, int C, int HW) {
  long idx = (long)blockIdx.x * blockDim.x + threadIdx.x;
  if (idx >= n) return;
  int c = (int)((idx / HW) % C);
  X[idx] = selu_f(X[idx] * g[c] * RSQ + bb[c]);
}

// ---------- e_S: softmax over W + weighted sum + pos_S (per b,c,h) ----------
__global__ void k_att_rows(const float* __restrict__ Hx, const float* __restrict__ Wm,
                           const float* __restrict__ posS, float* __restrict__ ES,
                           int B, int C, int H, int W) {
  int idx = blockIdx.x * blockDim.x + threadIdx.x;
  int n = B * C * H;
  if (idx >= n) return;
  int b = idx / (C * H); int rr = idx - b * C * H; int c = rr / H; int oh = rr - c * H;
  size_t base = (((size_t)b * C + c) * H + oh) * W;
  float mx = -1e30f;
  for (int w = 0; w < W; ++w) mx = fmaxf(mx, Wm[base + w]);
  float s = 0.f, acc = 0.f;
  for (int w = 0; w < W; ++w) { float e = expf(Wm[base + w] - mx); s += e; acc += e * Hx[base + w]; }
  ES[((size_t)b * H + oh) * C + c] = acc / s + posS[oh * C + c];
}

// ---------- e_T: softmax over H + weighted sum (per b,c,w) ------------------
__global__ void k_att_cols(const float* __restrict__ Hx, const float* __restrict__ Wm,
                           float* __restrict__ ET, int B, int C, int H, int W) {
  int idx = blockIdx.x * blockDim.x + threadIdx.x;
  int n = B * C * W;
  if (idx >= n) return;
  int b = idx / (C * W); int rr = idx - b * C * W; int c = rr / W; int ow = rr - c * W;
  size_t base = ((size_t)b * C + c) * (size_t)H * W + ow;
  float mx = -1e30f;
  for (int h = 0; h < H; ++h) mx = fmaxf(mx, Wm[base + (size_t)h * W]);
  float s = 0.f, acc = 0.f;
  for (int h = 0; h < H; ++h) { float e = expf(Wm[base + (size_t)h * W] - mx); s += e; acc += e * Hx[base + (size_t)h * W]; }
  ET[((size_t)b * W + ow) * C + c] = acc / s;
}

// ------------------------- small dense linear layer -------------------------
__global__ void k_lin(const float* __restrict__ X, const float* __restrict__ W,
                      const float* __restrict__ bias, float* __restrict__ Y,
                      int B, int K, int O) {
  int idx = blockIdx.x * blockDim.x + threadIdx.x;
  if (idx >= B * O) return;
  int b = idx / O, o2 = idx % O;
  float s = bias[o2];
  for (int k = 0; k < K; ++k) s += X[(size_t)b * K + k] * W[(size_t)k * O + o2];
  Y[(size_t)b * O + o2] = s;
}

// ----------------------------- GAT attention logits -------------------------
__global__ void k_gat_logits(const float* __restrict__ E, const float* __restrict__ Wp,
                             const float* __restrict__ bp, const float* __restrict__ aw,
                             float* __restrict__ BD, int B, int N, int Di, int Do,
                             float invtemp) {
  int idx = blockIdx.x * blockDim.x + threadIdx.x;
  if (idx >= B * N * N) return;
  int b = idx / (N * N); int rr = idx - b * N * N; int i = rr / N; int j = rr - i * N;
  float pm[32];
  const float* ei = E + ((size_t)b * N + i) * Di;
  const float* ej = E + ((size_t)b * N + j) * Di;
  for (int d = 0; d < Di; ++d) pm[d] = ei[d] * ej[d];
  float s = 0.f;
  for (int o2 = 0; o2 < Do; ++o2) {
    float t = bp[o2];
    for (int d = 0; d < Di; ++d) t += pm[d] * Wp[d * Do + o2];
    s += tanhf(t) * aw[o2];
  }
  BD[idx] = s * invtemp;
}

// --------------- GAT aggregation: softmax row, pw/pwo, BN, SELU -------------
__global__ void k_gat_agg(const float* __restrict__ E, const float* __restrict__ BD,
                          const float* __restrict__ pww, const float* __restrict__ pwb,
                          const float* __restrict__ pwow, const float* __restrict__ pwob,
                          const float* __restrict__ bng, const float* __restrict__ bnb,
                          float* __restrict__ Yo, int B, int N, int Di, int Do) {
  int idx = blockIdx.x * blockDim.x + threadIdx.x;
  if (idx >= B * N) return;
  int b = idx / N, i = idx % N;
  const float* row = BD + ((size_t)b * N + i) * N;
  float mx = -1e30f;
  for (int j = 0; j < N; ++j) mx = fmaxf(mx, row[j]);
  float s = 0.f, agg[32];
  for (int d = 0; d < Di; ++d) agg[d] = 0.f;
  for (int j = 0; j < N; ++j) {
    float e = expf(row[j] - mx); s += e;
    const float* ej = E + ((size_t)b * N + j) * Di;
    for (int d = 0; d < Di; ++d) agg[d] += e * ej[d];
  }
  float inv = 1.f / s;
  const float* ei = E + ((size_t)b * N + i) * Di;
  for (int o2 = 0; o2 < Do; ++o2) {
    float y = pwb[o2] + pwob[o2];
    for (int d = 0; d < Di; ++d) y += agg[d] * inv * pww[d * Do + o2] + ei[d] * pwow[d * Do + o2];
    y = selu_f(y * bng[o2] * RSQ + bnb[o2]);
    Yo[((size_t)b * N + i) * Do + o2] = y;
  }
}

// -------- graph_pool: sigmoid(lin) scores, torch-style top-k, gather --------
__global__ void k_pool(const float* __restrict__ Hx, const float* __restrict__ pw,
                       const float* __restrict__ pb, float* __restrict__ Yo,
                       int B, int N, int D, int k) {
  int b = blockIdx.x * blockDim.x + threadIdx.x;
  if (b >= B) return;
  float sc[67];
  bool used[67];
  for (int i = 0; i < N; ++i) {
    float s = pb[0];
    const float* hi = Hx + ((size_t)b * N + i) * D;
    for (int d = 0; d < D; ++d) s += hi[d] * pw[d];
    sc[i] = 1.f / (1.f + expf(-s));
    used[i] = false;
  }
  for (int t = 0; t < k; ++t) {
    int bi = 0; float best = -1e30f;
    for (int i = 0; i < N; ++i)
      if (!used[i] && sc[i] > best) { best = sc[i]; bi = i; }
    used[bi] = true;
    const float* hi = Hx + ((size_t)b * N + bi) * D;
    float* yo = Yo + ((size_t)b * k + t) * D;
    for (int d = 0; d < D; ++d) yo[d] = hi[d] * best;
  }
}

// ------------------------------- HTRG kernels -------------------------------
__global__ void k_htrg_x(const float* __restrict__ X1, const float* __restrict__ X2,
                         const float* __restrict__ w1, const float* __restrict__ b1,
                         const float* __restrict__ w2, const float* __restrict__ b2,
                         float* __restrict__ Xo, int B, int n1, int n2, int Di) {
  int N = n1 + n2;
  int idx = blockIdx.x * blockDim.x + threadIdx.x;
  if (idx >= B * N) return;
  int b = idx / N, i = idx % N;
  const float *src, *w, *bb2;
  if (i < n1) { src = X1 + ((size_t)b * n1 + i) * Di;        w = w1; bb2 = b1; }
  else        { src = X2 + ((size_t)b * n2 + (i - n1)) * Di; w = w2; bb2 = b2; }
  float xl[32];
  for (int d = 0; d < Di; ++d) xl[d] = src[d];
  float* dst = Xo + ((size_t)b * N + i) * Di;
  for (int o2 = 0; o2 < Di; ++o2) {
    float s = bb2[o2];
    for (int d = 0; d < Di; ++d) s += xl[d] * w[d * Di + o2];
    dst[o2] = s;
  }
}

__global__ void k_htrg_board(const float* __restrict__ Xo, const float* __restrict__ Wp,
                             const float* __restrict__ bp, const float* __restrict__ w11,
                             const float* __restrict__ w22, const float* __restrict__ w12,
                             float* __restrict__ BD, int B, int N, int n1,
                             int Di, int Do, float invtemp) {
  int idx = blockIdx.x * blockDim.x + threadIdx.x;
  if (idx >= B * N * N) return;
  int b = idx / (N * N); int rr = idx - b * N * N; int i = rr / N; int j = rr - i * N;
  const float* wi = (i < n1 && j < n1) ? w11 : ((i >= n1 && j >= n1) ? w22 : w12);
  float pm[32];
  const float* xi = Xo + ((size_t)b * N + i) * Di;
  const float* xj = Xo + ((size_t)b * N + j) * Di;
  for (int d = 0; d < Di; ++d) pm[d] = xi[d] * xj[d];
  float s = 0.f;
  for (int o2 = 0; o2 < Do; ++o2) {
    float t = bp[o2];
    for (int d = 0; d < Di; ++d) t += pm[d] * Wp[d * Do + o2];
    s += tanhf(t) * wi[o2];
  }
  BD[idx] = s * invtemp;
}

__global__ void k_htrg_attM(const float* __restrict__ Xo, const float* __restrict__ Mi,
                            int mstride, const float* __restrict__ Wp,
                            const float* __restrict__ bp, const float* __restrict__ wM,
                            float* __restrict__ AM, int B, int N, int Di, int Do,
                            float invtemp) {
  int idx = blockIdx.x * blockDim.x + threadIdx.x;
  if (idx >= B * N) return;
  int b = idx / N, i = idx % N;
  const float* xi = Xo + ((size_t)b * N + i) * Di;
  const float* mm = Mi + (size_t)b * mstride;
  float pm[32];
  for (int d = 0; d < Di; ++d) pm[d] = xi[d] * mm[d];
  float s = 0.f;
  for (int o2 = 0; o2 < Do; ++o2) {
    float t = bp[o2];
    for (int d = 0; d < Di; ++d) t += pm[d] * Wp[d * Do + o2];
    s += tanhf(t) * wM[o2];
  }
  AM[idx] = s * invtemp;
}

__global__ void k_htrg_master(const float* __restrict__ Xo, const float* __restrict__ AM,
                              const float* __restrict__ Mi, int mstride,
                              const float* __restrict__ pwMw, const float* __restrict__ pwMb,
                              const float* __restrict__ pwoMw, const float* __restrict__ pwoMb,
                              float* __restrict__ Mo, int accum,
                              int B, int N, int Di, int Do) {
  int b = blockIdx.x * blockDim.x + threadIdx.x;
  if (b >= B) return;
  const float* row = AM + (size_t)b * N;
  float mx = -1e30f;
  for (int i = 0; i < N; ++i) mx = fmaxf(mx, row[i]);
  float s = 0.f, agg[32];
  for (int d = 0; d < Di; ++d) agg[d] = 0.f;
  for (int i = 0; i < N; ++i) {
    float e = expf(row[i] - mx); s += e;
    const float* xi = Xo + ((size_t)b * N + i) * Di;
    for (int d = 0; d < Di; ++d) agg[d] += e * xi[d];
  }
  float inv = 1.f / s;
  float ml[32];
  for (int d = 0; d < Di; ++d) ml[d] = Mi[(size_t)b * mstride + d];
  for (int o2 = 0; o2 < Do; ++o2) {
    float y = pwMb[o2] + pwoMb[o2];
    for (int d = 0; d < Di; ++d) y += agg[d] * inv * pwMw[d * Do + o2] + ml[d] * pwoMw[d * Do + o2];
    if (accum) Mo[(size_t)b * Do + o2] += y; else Mo[(size_t)b * Do + o2] = y;
  }
}

__global__ void k_htrg_y(const float* __restrict__ Xo, const float* __restrict__ BD,
                         const float* __restrict__ pww, const float* __restrict__ pwb,
                         const float* __restrict__ pwow, const float* __restrict__ pwob,
                         const float* __restrict__ bng, const float* __restrict__ bnb,
                         float* __restrict__ O1, float* __restrict__ O2, int accum,
                         int B, int N, int n1, int Di, int Do) {
  int idx = blockIdx.x * blockDim.x + threadIdx.x;
  if (idx >= B * N) return;
  int b = idx / N, i = idx % N;
  int n2 = N - n1;
  const float* row = BD + ((size_t)b * N + i) * N;
  float mx = -1e30f;
  for (int j = 0; j < N; ++j) mx = fmaxf(mx, row[j]);
  float s = 0.f, agg[32];
  for (int d = 0; d < Di; ++d) agg[d] = 0.f;
  for (int j = 0; j < N; ++j) {
    float e = expf(row[j] - mx); s += e;
    const float* xj = Xo + ((size_t)b * N + j) * Di;
    for (int d = 0; d < Di; ++d) agg[d] += e * xj[d];
  }
  float inv = 1.f / s;
  const float* xi = Xo + ((size_t)b * N + i) * Di;
  float* dst = (i < n1) ? (O1 + ((size_t)b * n1 + i) * Do)
                        : (O2 + ((size_t)b * n2 + (i - n1)) * Do);
  for (int o2 = 0; o2 < Do; ++o2) {
    float y = pwb[o2] + pwob[o2];
    for (int d = 0; d < Di; ++d) y += agg[d] * inv * pww[d * Do + o2] + xi[d] * pwow[d * Do + o2];
    y = selu_f(y * bng[o2] * RSQ + bnb[o2]);
    if (accum) dst[o2] += y; else dst[o2] = y;
  }
}

// ------------------------------- misc kernels -------------------------------
__global__ void k_eltmax(const float* __restrict__ A, const float* __restrict__ Bp,
                         float* __restrict__ C, long n) {
  long idx = (long)blockIdx.x * blockDim.x + threadIdx.x;
  if (idx >= n) return;
  C[idx] = fmaxf(A[idx], Bp[idx]);
}

__global__ void k_last(const float* __restrict__ OT, const float* __restrict__ OS,
                       const float* __restrict__ MXp, float* __restrict__ L, int B) {
  int idx = blockIdx.x * blockDim.x + threadIdx.x;
  if (idx >= B * 32) return;
  int b = idx / 32, d = idx % 32;
  float maT = -1e30f, mnT = 0.f;
  for (int i = 0; i < 23; ++i) { float v = OT[((size_t)b * 23 + i) * 32 + d]; maT = fmaxf(maT, fabsf(v)); mnT += v; }
  mnT /= 23.f;
  float maS = -1e30f, mnS = 0.f;
  for (int i = 0; i < 11; ++i) { float v = OS[((size_t)b * 11 + i) * 32 + d]; maS = fmaxf(maS, fabsf(v)); mnS += v; }
  mnS /= 11.f;
  L[b * 160 + d]       = maT;
  L[b * 160 + 32 + d]  = mnT;
  L[b * 160 + 64 + d]  = maS;
  L[b * 160 + 96 + d]  = mnS;
  L[b * 160 + 128 + d] = MXp[b * 32 + d];
}

// ---------------------------------------------------------------------------
static inline long cdivl(long a, long b) { return (a + b - 1) / b; }

extern "C" void kernel_launch(void* const* d_in, const int* in_sizes, int n_in,
                              void* d_out, int out_size, void* d_ws, size_t ws_size,
                              hipStream_t stream) {
  (void)in_sizes; (void)n_in; (void)out_size; (void)ws_size;
  auto P = [&](int i) { return (const float*)d_in[i]; };
  float* out = (float*)d_out;
  float* ws  = (float*)d_ws;

  const int Bz = 128, T = 201, Dssl = 768, NLL = 128;
  const int H = 42, Wd = 67;

  // d_out flat offsets (return-tuple order)
  const long OFF_OUT = 0, OFF_SPEC = 256, OFF_TEMP = 512, OFF_GS = 768, OFF_GT = 1024,
             OFF_HSS = 1280, OFF_HST = 1536, OFF_MF1 = 1792, OFF_MF2 = 46848,
             OFF_FF1 = 141056, OFF_FF2 = 186112, OFF_XS = 280320;

  // ---------------- workspace layout (floats), ~266 MB ----------------
  size_t o = 0;
  float* POOL0 = ws + o; o += (size_t)Bz * H * Wd;              // (B,1,42,67)
  float* R     = ws + o; o += 46400000ULL;                      // conv ping-pong / wmap
  float* Hbuf  = ws + o; o += (size_t)Bz * 24 * H * Wd;         // h / h_act
  float* W24   = ws + o; o += (size_t)Bz * 24 * H * Wd;         // attention map w
  float* ES    = ws + o; o += (size_t)Bz * 42 * 24;
  float* ET    = ws + o; o += (size_t)Bz * 67 * 24;
  float* GS    = ws + o; o += (size_t)Bz * 42 * 24;
  float* GT    = ws + o; o += (size_t)Bz * 67 * 24;
  float* OUTS  = ws + o; o += (size_t)Bz * 16 * 24;
  float* OUTT  = ws + o; o += (size_t)Bz * 33 * 24;
  float* BOARD = ws + o; o += (size_t)Bz * 67 * 67;
  float* ATTM  = ws + o; o += (size_t)Bz * 64;
  float* XH    = ws + o; o += (size_t)Bz * 49 * 32;
  float* T1a = ws + o; o += (size_t)Bz * 33 * 32;
  float* S1a = ws + o; o += (size_t)Bz * 16 * 32;
  float* M1  = ws + o; o += (size_t)Bz * 32;
  float* T1  = ws + o; o += (size_t)Bz * 23 * 32;
  float* S1  = ws + o; o += (size_t)Bz * 11 * 32;
  float* T2a = ws + o; o += (size_t)Bz * 33 * 32;
  float* S2a = ws + o; o += (size_t)Bz * 16 * 32;
  float* M2  = ws + o; o += (size_t)Bz * 32;
  float* T2  = ws + o; o += (size_t)Bz * 23 * 32;
  float* S2  = ws + o; o += (size_t)Bz * 11 * 32;
  float* OT  = ws + o; o += (size_t)Bz * 23 * 32;
  float* OS  = ws + o; o += (size_t)Bz * 11 * 32;
  float* MX  = ws + o; o += (size_t)Bz * 32;
  float* LAST = ws + o; o += (size_t)Bz * 160;
  _Float16* WT16 = (_Float16*)(ws + o); o += 49152;   // LL weight f16 (N=128,K=768)
  _Float16* WPAD = (_Float16*)(ws + o); o += 4096;    // conv weight pad (<=8192 halfs)
  int4*     TAB  = (int4*)(ws + o);     o += 1024;    // im2col table (<=256 int4)

  // ---------------- 1) xs = x @ LL.w + LL.b  (WMMA GEMM) ----------------
  float* XS = out + OFF_XS;
  {
    int M = Bz * T, N = NLL, K = Dssl;
    k_prep_wt<<<(unsigned)cdivl((long)K * N, 256), 256, 0, stream>>>(P(1), WT16, K, N);
    long waves = (long)(M >> 4) * (N >> 4);
    k_gemm<<<(unsigned)cdivl(waves * 32, 256), 256, 0, stream>>>(P(0), WT16, P(2), XS, M, N, K);
  }

  // ---------------- 2) maxpool 3x3/3 + first_bn + selu ----------------
  k_pool_bnselu<<<(unsigned)cdivl((long)Bz * H * Wd, 256), 256, 0, stream>>>(
      XS, P(3), P(4), POOL0, Bz, H, Wd, T, NLL);

  // ---------------- 3) encoder residual blocks (WMMA conv) ----------------
  auto conv = [&](const float* X, const float* Wt, const float* bias, const float* addend,
                  const float* pres, const float* preb, const float* posts, const float* postb,
                  float* Y, int Ci, int Hi, int Wi, int Co, int Ho, int Wo,
                  int kh, int kw, int ph, int pw) {
    int K = Ci * kh * kw;
    int K32 = (K + 31) & ~31;
    int Co16 = (Co + 15) & ~15;
    k_prep_w<<<(unsigned)cdivl((long)Co16 * K32, 256), 256, 0, stream>>>(Wt, WPAD, Co, K, Co16, K32);
    k_prep_tab<<<1, 256, 0, stream>>>(TAB, K, K32, kh, kw, ph, pw);
    int M = Bz * Ho * Wo;                         // multiple of 16 (B=128)
    long waves = (long)(M >> 4) * (Co16 >> 4);
    k_conv<<<(unsigned)cdivl(waves * 32, 256), 256, 0, stream>>>(
        X, WPAD, TAB, bias, addend, pres, preb, posts, postb, Y,
        M, Ci, Hi, Wi, Co, Ho, Wo, K, K32);
  };

  struct Blk { int base, ci, co; bool bn1, down; };
  const Blk blks[6] = { {7, 1, 32, false, true}, {15, 32, 32, true, false},
                        {23, 32, 24, true, true}, {33, 24, 24, true, false},
                        {41, 24, 24, true, false}, {49, 24, 24, true, false} };
  float* ping = R;
  float* pong = R + 11526144;   // B*32*42*67
  float* idn  = R + 23052288;
  float* tmp  = R + 34578432;   // B*32*43*67
  float* cur  = POOL0;
  for (int i = 0; i < 6; ++i) {
    const Blk& bk = blks[i];
    const float* bn1g = bk.bn1 ? P(bk.base + 6) : nullptr;
    const float* bn1b = bk.bn1 ? P(bk.base + 7) : nullptr;
    int dbase = bk.base + 6 + (bk.bn1 ? 2 : 0);
    float* outp = (i == 5) ? Hbuf : ping;
    // conv1: k(2,3) pad(1,1): H -> H+1, pre = selu(bn1(x)) if present
    conv(cur, P(bk.base + 0), P(bk.base + 1), nullptr, bn1g, bn1b, nullptr, nullptr,
         tmp, bk.ci, H, Wd, bk.co, H + 1, Wd, 2, 3, 1, 1);
    // identity path
    const float* idptr = cur;
    if (bk.down) {
      conv(cur, P(dbase), P(dbase + 1), nullptr, nullptr, nullptr, nullptr, nullptr,
           idn, bk.ci, H, Wd, bk.co, H, Wd, 1, 3, 0, 1);
      idptr = idn;
    }
    // conv2: k(2,3) pad(0,1): H+1 -> H, pre = selu(bn2(.)), addend = identity
    conv(tmp, P(bk.base + 4), P(bk.base + 5), idptr, P(bk.base + 2), P(bk.base + 3),
         nullptr, nullptr, outp, bk.co, H + 1, Wd, bk.co, H, Wd, 2, 3, 0, 1);
    cur = outp;
    float* t2 = ping; ping = pong; pong = t2;
  }

  // ---------------- 4) first_bn1 + selu (in place) ----------------
  k_bnselu<<<(unsigned)cdivl((long)Bz * 24 * H * Wd, 256), 256, 0, stream>>>(
      Hbuf, P(5), P(6), (long)Bz * 24 * H * Wd, 24, H * Wd);

  // ---------------- 5) attention convs (1x1, WMMA) ----------------
  float* WMAP = R;  // conv scratch region reused: (B,128,42,67)
  conv(Hbuf, P(57), P(58), nullptr, nullptr, nullptr, P(59), P(60),
       WMAP, 24, H, Wd, 128, H, Wd, 1, 1, 0, 0);            // post: bn(selu(.))
  conv(WMAP, P(61), P(62), nullptr, nullptr, nullptr, nullptr, nullptr,
       W24, 128, H, Wd, 24, H, Wd, 1, 1, 0, 0);

  // ---------------- 6) e_S / e_T ----------------
  k_att_rows<<<(unsigned)cdivl((long)Bz * 24 * H, 256), 256, 0, stream>>>(
      Hbuf, W24, P(63), ES, Bz, 24, H, Wd);
  k_att_cols<<<(unsigned)cdivl((long)Bz * 24 * Wd, 256), 256, 0, stream>>>(
      Hbuf, W24, ET, Bz, 24, H, Wd);

  // spectral / temporal heads
  k_lin<<<(unsigned)cdivl(Bz * 2, 128), 128, 0, stream>>>(ES, P(184), P(185), out + OFF_SPEC, Bz, 1008, 2);
  k_lin<<<(unsigned)cdivl(Bz * 2, 128), 128, 0, stream>>>(ET, P(186), P(187), out + OFF_TEMP, Bz, 1608, 2);

  // ---------------- 7) GAT + graph pools ----------------
  k_gat_logits<<<(unsigned)cdivl((long)Bz * 42 * 42, 256), 256, 0, stream>>>(
      ES, P(66), P(67), P(68), BOARD, Bz, 42, 24, 24, 0.5f);
  k_gat_agg<<<(unsigned)cdivl(Bz * 42, 128), 128, 0, stream>>>(
      ES, BOARD, P(69), P(70), P(71), P(72), P(73), P(74), GS, Bz, 42, 24, 24);
  k_pool<<<1, 128, 0, stream>>>(GS, P(172), P(173), OUTS, Bz, 42, 24, 16);

  k_gat_logits<<<(unsigned)cdivl((long)Bz * 67 * 67, 256), 256, 0, stream>>>(
      ET, P(75), P(76), P(77), BOARD, Bz, 67, 24, 24, 0.5f);
  k_gat_agg<<<(unsigned)cdivl(Bz * 67, 128), 128, 0, stream>>>(
      ET, BOARD, P(78), P(79), P(80), P(81), P(82), P(83), GT, Bz, 67, 24, 24);
  k_pool<<<1, 128, 0, stream>>>(GT, P(174), P(175), OUTT, Bz, 67, 24, 33);

  // ---------------- 8) HTRG stacks ----------------
  auto htrg = [&](int base, const float* x1, int n1, const float* x2, int n2,
                  int Di, int Do, const float* mIn, int mstride, float* mOut, int maccum,
                  float* o1, float* o2, int yaccum) {
    int N = n1 + n2;
    k_htrg_x<<<(unsigned)cdivl(Bz * N, 128), 128, 0, stream>>>(
        x1, x2, P(base + 0), P(base + 1), P(base + 2), P(base + 3), XH, Bz, n1, n2, Di);
    k_htrg_board<<<(unsigned)cdivl((long)Bz * N * N, 256), 256, 0, stream>>>(
        XH, P(base + 4), P(base + 5), P(base + 18), P(base + 19), P(base + 20),
        BOARD, Bz, N, n1, Di, Do, 0.01f);
    k_htrg_attM<<<(unsigned)cdivl(Bz * N, 128), 128, 0, stream>>>(
        XH, mIn, mstride, P(base + 6), P(base + 7), P(base + 21), ATTM, Bz, N, Di, Do, 0.01f);
    k_htrg_master<<<1, 128, 0, stream>>>(
        XH, ATTM, mIn, mstride, P(base + 12), P(base + 13), P(base + 14), P(base + 15),
        mOut, maccum, Bz, N, Di, Do);
    k_htrg_y<<<(unsigned)cdivl(Bz * N, 128), 128, 0, stream>>>(
        XH, BOARD, P(base + 8), P(base + 9), P(base + 10), P(base + 11),
        P(base + 16), P(base + 17), o1, o2, yaccum, Bz, N, n1, Di, Do);
  };

  // branch 1
  htrg(84, OUTT, 33, OUTS, 16, 24, 32, P(64), 0, M1, 0, T1a, S1a, 0);
  k_pool<<<1, 128, 0, stream>>>(S1a, P(176), P(177), S1, Bz, 16, 32, 11);
  k_pool<<<1, 128, 0, stream>>>(T1a, P(178), P(179), T1, Bz, 33, 32, 23);
  hipMemcpyAsync(out + OFF_MF1, S1, (size_t)Bz * 352 * sizeof(float), hipMemcpyDeviceToDevice, stream);
  hipMemcpyAsync(out + OFF_MF2, T1, (size_t)Bz * 736 * sizeof(float), hipMemcpyDeviceToDevice, stream);
  k_lin<<<(unsigned)cdivl(Bz * 2, 128), 128, 0, stream>>>(S1, P(188), P(189), out + OFF_GS, Bz, 352, 2);
  k_lin<<<(unsigned)cdivl(Bz * 2, 128), 128, 0, stream>>>(T1, P(190), P(191), out + OFF_GT, Bz, 736, 2);
  htrg(106, T1, 23, S1, 11, 32, 32, M1, 32, M1, 1, T1, S1, 1);

  // branch 2
  htrg(128, OUTT, 33, OUTS, 16, 24, 32, P(65), 0, M2, 0, T2a, S2a, 0);
  k_pool<<<1, 128, 0, stream>>>(S2a, P(180), P(181), S2, Bz, 16, 32, 11);
  k_pool<<<1, 128, 0, stream>>>(T2a, P(182), P(183), T2, Bz, 33, 32, 23);
  htrg(150, T2, 23, S2, 11, 32, 32, M2, 32, M2, 1, T2, S2, 1);
  hipMemcpyAsync(out + OFF_FF1, S2, (size_t)Bz * 352 * sizeof(float), hipMemcpyDeviceToDevice, stream);
  hipMemcpyAsync(out + OFF_FF2, T2, (size_t)Bz * 736 * sizeof(float), hipMemcpyDeviceToDevice, stream);
  k_lin<<<(unsigned)cdivl(Bz * 2, 128), 128, 0, stream>>>(S2, P(192), P(193), out + OFF_HSS, Bz, 352, 2);
  k_lin<<<(unsigned)cdivl(Bz * 2, 128), 128, 0, stream>>>(T2, P(194), P(195), out + OFF_HST, Bz, 736, 2);

  // ---------------- 9) final fusion + output head ----------------
  k_eltmax<<<(unsigned)cdivl((long)Bz * 736, 256), 256, 0, stream>>>(T1, T2, OT, (long)Bz * 736);
  k_eltmax<<<(unsigned)cdivl((long)Bz * 352, 256), 256, 0, stream>>>(S1, S2, OS, (long)Bz * 352);
  k_eltmax<<<(unsigned)cdivl((long)Bz * 32, 256), 256, 0, stream>>>(M1, M2, MX, (long)Bz * 32);
  k_last<<<(unsigned)cdivl(Bz * 32, 128), 128, 0, stream>>>(OT, OS, MX, LAST, Bz);
  k_lin<<<(unsigned)cdivl(Bz * 2, 128), 128, 0, stream>>>(LAST, P(196), P(197), out + OFF_OUT, Bz, 160, 2);
}
